// GNN_41506563948780
// MI455X (gfx1250) — matrix operand order repaired and checked by reading, not compiled, not run on hardware
//
#include <hip/hip_runtime.h>
#include <hip/hip_bf16.h>
#include <stdint.h>

typedef unsigned short u16;
typedef unsigned int   u32;
typedef unsigned long long u64;

typedef __attribute__((ext_vector_type(16))) __bf16 v16bf;
typedef __attribute__((ext_vector_type(8)))  float  v8f;

union V16BF {
    v16bf v;
    uint4 q[2];
    u16   u[16];
};

__device__ __forceinline__ u16 f2bf(float x) {
    union { float f; u32 u; } c; c.f = x;
    u32 r = c.u + 0x7FFFu + ((c.u >> 16) & 1u);   // round-to-nearest-even
    return (u16)(r >> 16);
}
__device__ __forceinline__ float bf2f(u16 h) {
    union { float f; u32 u; } c; c.u = ((u32)h) << 16;
    return c.f;
}

// Load one A fragment's worth of fp32 data (16 contiguous values K=c..c+15
// in contiguous k order, matching the B fragments) and convert to bf16.
__device__ __forceinline__ void load_f32A(const float* __restrict__ s, V16BF& a) {
    const float4* q0 = (const float4*)s;
    const float4* q1 = (const float4*)(s + 8);
    float4 v0 = q0[0], v1 = q0[1], v2 = q1[0], v3 = q1[1];
    a.u[0]  = f2bf(v0.x); a.u[1]  = f2bf(v0.y); a.u[2]  = f2bf(v0.z); a.u[3]  = f2bf(v0.w);
    a.u[4]  = f2bf(v1.x); a.u[5]  = f2bf(v1.y); a.u[6]  = f2bf(v1.z); a.u[7]  = f2bf(v1.w);
    a.u[8]  = f2bf(v2.x); a.u[9]  = f2bf(v2.y); a.u[10] = f2bf(v2.z); a.u[11] = f2bf(v2.w);
    a.u[12] = f2bf(v3.x); a.u[13] = f2bf(v3.y); a.u[14] = f2bf(v3.z); a.u[15] = f2bf(v3.w);
}

// Async-copy one 64B weight row (both LDS and global sides advance with the
// instruction offset, ISA 08_async_tensor §4.4). Tracked with ASYNCcnt.
__device__ __forceinline__ void async_stage_row(u32 lds_off, const u16* gsrc) {
    u64 ga = (u64)(uintptr_t)gsrc;
    asm volatile(
        "global_load_async_to_lds_b128 %0, %1, off\n\t"
        "global_load_async_to_lds_b128 %0, %1, off offset:16\n\t"
        "global_load_async_to_lds_b128 %0, %1, off offset:32\n\t"
        "global_load_async_to_lds_b128 %0, %1, off offset:48"
        :: "v"(lds_off), "v"(ga) : "memory");
}
__device__ __forceinline__ void wait_async0() {
    asm volatile("s_wait_asynccnt 0x0" ::: "memory");
}

#define LN_EPS 1e-5f

// ---------------------------------------------------------------------------
// Fused gather + GEMM + bias + LayerNorm + ReLU.
// MODE 0: edge MLP   A = [edges | nodes[snd] | nodes[rcv] | globals]
// MODE 1: node MLP   A = [nodes | sent(f32)  | recv(f32)  | globals]
// MODE 2: glob MLP   A = [n_attr(f32) | e_attr(f32) | globals(f32)]
// Block = 8 waves (256 thr): 4 M-tiles x 2 N-halves; each wave = 8 N-tiles.
// Weight slabs double-buffered in LDS via global_load_async_to_lds_b128
// (no staging VGPRs); statically-indexed 2-phase k-loop; gathered A fragment
// prefetched one phase ahead; ping-pong B fragments from LDS.
// ---------------------------------------------------------------------------
template <int MODE>
__global__ __launch_bounds__(256) void gn_mlp_wmma(
    int in_dim, int gdim, int log_pg,
    const u16* __restrict__ seg0, const u16* __restrict__ gnodes,
    const int* __restrict__ snd, const int* __restrict__ rcv,
    const float* __restrict__ f1, const float* __restrict__ f2,
    const float* __restrict__ gvec,
    const u16* __restrict__ wt,
    const float* __restrict__ bias, const float* __restrict__ gamma,
    const float* __restrict__ beta,
    u16* __restrict__ out_bf, float* __restrict__ out_f)
{
    // two 32-K B slabs, rows padded to 80B (stride/16 odd -> minimal conflicts)
    __shared__ __align__(16) u16 Bs[2][256 * 40];
    __shared__ float red1[8][2][8];   // [wave][half][row] partial row sums
    __shared__ float red2[8][2][8];

    const int lane = threadIdx.x & 31;
    const int wv   = threadIdx.x >> 5;   // 0..7
    const int h    = lane >> 4;
    const int l16  = lane & 15;
    const int mt   = wv >> 1;            // M-tile within block
    const int nh   = wv & 1;             // N half: tiles 8*nh .. 8*nh+7
    const int m0   = (blockIdx.x * 4 + mt) * 16;
    const int row  = m0 + l16;

    const u16   *p0 = nullptr, *p1 = nullptr, *p2 = nullptr;
    const float *pf1 = nullptr, *pf2 = nullptr, *pg = nullptr;
    if (MODE == 0) {
        p0 = seg0   + (size_t)row * 256;
        p1 = gnodes + (size_t)snd[row] * 256;
        p2 = gnodes + (size_t)rcv[row] * 256;
        pg = gvec   + (size_t)(row >> log_pg) * gdim;
    } else if (MODE == 1) {
        p0  = seg0 + (size_t)row * 256;
        pf1 = f1   + (size_t)row * 256;
        pf2 = f2   + (size_t)row * 256;
        pg  = gvec + (size_t)(row >> log_pg) * gdim;
    } else {
        pf1 = f1   + (size_t)row * 256;
        pf2 = f2   + (size_t)row * 256;
        pg  = gvec + (size_t)row * gdim;
    }

    v8f acc[8];
#pragma unroll
    for (int t = 0; t < 8; ++t)
#pragma unroll
        for (int r = 0; r < 8; ++r) acc[t][r] = 0.0f;

    const int lbase = l16 * 40 + 16 * h + nh * 8 * 640;  // per-lane B offset in slab
    const int nk = in_dim >> 5;                          // always even here

    // per-thread staging addresses (one 64B weight row per thread per slab)
    const u16* grow = wt + (size_t)threadIdx.x * in_dim;
    const u32 lds0 = (u32)(uintptr_t)&Bs[0][threadIdx.x * 40];
    const u32 lds1 = (u32)(uintptr_t)&Bs[1][threadIdx.x * 40];

    auto load_a = [&](int k, V16BF& a) {
        const int c0 = k + 16 * h;   // first 8-elem chunk; second at +8
        if (MODE == 2) {
            const float* s = (k < 256) ? (pf1 + c0)
                           : (k < 512) ? (pf2 + (c0 - 256))
                                       : (pg  + (c0 - 512));
            load_f32A(s, a);
        } else if (MODE == 0) {
            if (k < 768) {
                const u16* s = (k < 256) ? (p0 + c0)
                             : (k < 512) ? (p1 + (c0 - 256))
                                         : (p2 + (c0 - 512));
                a.q[0] = *(const uint4*)s;
                a.q[1] = *(const uint4*)(s + 8);
            } else {
                load_f32A(pg + (c0 - 768), a);
            }
        } else {
            if (k < 256) {
                const u16* s = p0 + c0;
                a.q[0] = *(const uint4*)s;
                a.q[1] = *(const uint4*)(s + 8);
            } else {
                const float* s = (k < 512) ? (pf1 + (c0 - 256))
                               : (k < 768) ? (pf2 + (c0 - 512))
                                           : (pg  + (c0 - 768));
                load_f32A(s, a);
            }
        }
    };
    auto compute = [&](const u16* bb, V16BF& a) {
        V16BF b0, b1;
        { const uint4* l = (const uint4*)bb; b0.q[0] = l[0]; b0.q[1] = l[1]; }
#pragma unroll
        for (int t = 0; t < 8; t += 2) {
            { const uint4* l = (const uint4*)(bb + (t + 1) * 640);
              b1.q[0] = l[0]; b1.q[1] = l[1]; }
            acc[t] = __builtin_amdgcn_wmma_f32_16x16x32_bf16(
                false, a.v, false, b0.v, (short)0, acc[t], false, false);
            if (t + 2 < 8) {
                const uint4* l = (const uint4*)(bb + (t + 2) * 640);
                b0.q[0] = l[0]; b0.q[1] = l[1];
            }
            acc[t + 1] = __builtin_amdgcn_wmma_f32_16x16x32_bf16(
                false, a.v, false, b1.v, (short)0, acc[t + 1], false, false);
        }
    };

    // prologue: async-stage slab 0, prefetch A(0)
    V16BF a0, a1;
    async_stage_row(lds0, grow);
    load_a(0, a0);
    wait_async0();
    __syncthreads();

    for (int ki = 0; ki < nk; ki += 2) {
        const int k0 = ki << 5;
        // phase 0: compute on Bs[0]/a0; async-stage slab(k0+32)->Bs[1], A(k0+32)
        async_stage_row(lds1, grow + k0 + 32);
        load_a(k0 + 32, a1);
        compute(&Bs[0][lbase], a0);
        wait_async0();
        __syncthreads();
        // phase 1: compute on Bs[1]/a1; async-stage slab(k0+64)->Bs[0], A(k0+64)
        const bool more = (ki + 2) < nk;
        if (more) {
            async_stage_row(lds0, grow + k0 + 64);
            load_a(k0 + 64, a0);
        }
        compute(&Bs[1][lbase], a1);
        wait_async0();
        __syncthreads();
    }

    // --- epilogue: + bias, LayerNorm over 256 cols (cross-wave), ReLU ---
    const int tt0 = nh * 8;
    float bl[8], gl[8], el[8];
#pragma unroll
    for (int t = 0; t < 8; ++t) {
        int n = 16 * (tt0 + t) + l16;
        bl[t] = bias[n]; gl[t] = gamma[n]; el[t] = beta[n];
    }
    float s1[8], s2[8];
#pragma unroll
    for (int r = 0; r < 8; ++r) { s1[r] = 0.f; s2[r] = 0.f; }
#pragma unroll
    for (int t = 0; t < 8; ++t)
#pragma unroll
        for (int r = 0; r < 8; ++r) {
            float x = acc[t][r] + bl[t];
            acc[t][r] = x;
            s1[r] += x; s2[r] += x * x;
        }
    // intra-wave: rows live within a 16-lane half; masks 1..8 stay inside it
#pragma unroll
    for (int m = 1; m <= 8; m <<= 1)
#pragma unroll
        for (int r = 0; r < 8; ++r) {
            s1[r] += __shfl_xor(s1[r], m, 32);
            s2[r] += __shfl_xor(s2[r], m, 32);
        }
    // cross-wave: combine with partner N-half (wave wv^1) via LDS
    if (l16 == 0) {
#pragma unroll
        for (int r = 0; r < 8; ++r) {
            red1[wv][h][r] = s1[r];
            red2[wv][h][r] = s2[r];
        }
    }
    __syncthreads();
#pragma unroll
    for (int r = 0; r < 8; ++r) {
        s1[r] += red1[wv ^ 1][h][r];
        s2[r] += red2[wv ^ 1][h][r];
    }
    float mu[8], rs[8];
#pragma unroll
    for (int r = 0; r < 8; ++r) {
        mu[r] = s1[r] * (1.0f / 256.0f);
        float var = s2[r] * (1.0f / 256.0f) - mu[r] * mu[r];
        rs[r] = rsqrtf(var + LN_EPS);
    }
#pragma unroll
    for (int t = 0; t < 8; ++t)
#pragma unroll
        for (int r = 0; r < 8; ++r) {
            float y = (acc[t][r] - mu[r]) * rs[r] * gl[t] + el[t];
            y = fmaxf(y, 0.0f);
            size_t o = (size_t)(m0 + r + 8 * h) * 256 + 16 * (tt0 + t) + l16;
            if (MODE == 2) out_f[o] = y; else out_bf[o] = f2bf(y);
        }
}

// ---------------------------------------------------------------------------
// Decoder: out[128,128] = globals[128,256] @ dec_w[256,128] + dec_b, fp32 out.
// One wave per 16-row M-tile, 8 N-tiles, B straight from L2 (tiny).
// ---------------------------------------------------------------------------
__global__ __launch_bounds__(32) void dec_wmma(
    const float* __restrict__ g, const u16* __restrict__ wt,  // wt: [128][256] bf16
    const float* __restrict__ bias, float* __restrict__ out)
{
    const int lane = threadIdx.x & 31;
    const int h = lane >> 4, l16 = lane & 15;
    const int m0 = blockIdx.x * 16;
    const float* pr = g + (size_t)(m0 + l16) * 256;

    v8f acc[8];
#pragma unroll
    for (int t = 0; t < 8; ++t)
#pragma unroll
        for (int r = 0; r < 8; ++r) acc[t][r] = 0.0f;

    for (int k = 0; k < 256; k += 32) {
        V16BF a;
        load_f32A(pr + k + 16 * h, a);
#pragma unroll
        for (int t = 0; t < 8; t += 2) {
            V16BF b0, b1;
            const u16* s0 = wt + (size_t)(16 * (t + 0) + l16) * 256 + k + 16 * h;
            const u16* s1 = wt + (size_t)(16 * (t + 1) + l16) * 256 + k + 16 * h;
            b0.q[0] = *(const uint4*)s0; b0.q[1] = *(const uint4*)(s0 + 8);
            b1.q[0] = *(const uint4*)s1; b1.q[1] = *(const uint4*)(s1 + 8);
            acc[t + 0] = __builtin_amdgcn_wmma_f32_16x16x32_bf16(
                false, a.v, false, b0.v, (short)0, acc[t + 0], false, false);
            acc[t + 1] = __builtin_amdgcn_wmma_f32_16x16x32_bf16(
                false, a.v, false, b1.v, (short)0, acc[t + 1], false, false);
        }
    }
#pragma unroll
    for (int t = 0; t < 8; ++t)
#pragma unroll
        for (int r = 0; r < 8; ++r)
            out[(size_t)(m0 + r + 8 * h) * 128 + 16 * t + l16] =
                acc[t][r] + bias[16 * t + l16];
}

// ---------------------------------------------------------------------------
// Small helper kernels
// ---------------------------------------------------------------------------
__global__ void embed_kernel(const float* __restrict__ x, const float* __restrict__ w,
                             const float* __restrict__ b, u16* __restrict__ out, int K)
{
    size_t tid = (size_t)blockIdx.x * blockDim.x + threadIdx.x;
    size_t row = tid >> 8;
    int col = (int)(tid & 255);
    const float* xr = x + row * K;
    float s = b[col];
    for (int k = 0; k < K; ++k) s += xr[k] * w[(size_t)k * 256 + col];
    out[row * 256 + col] = f2bf(s);
}

__global__ void prep_wt_kernel(const float* __restrict__ w, u16* __restrict__ wt,
                               int in_dim, int ncols)
{
    size_t i = (size_t)blockIdx.x * blockDim.x + threadIdx.x;
    if (i >= (size_t)in_dim * ncols) return;
    int n  = (int)(i / in_dim);
    int kk = (int)(i % in_dim);
    wt[i] = f2bf(w[(size_t)kk * ncols + n]);
}

__global__ void zero_f32(float* __restrict__ p, size_t n)
{
    size_t i = (size_t)blockIdx.x * blockDim.x + threadIdx.x;
    if (i < n) p[i] = 0.0f;
}

__global__ void scatter_kernel(const u16* __restrict__ ed, const int* __restrict__ snd,
                               const int* __restrict__ rcv, float* __restrict__ sent,
                               float* __restrict__ recv)
{
    size_t tid = (size_t)blockIdx.x * blockDim.x + threadIdx.x;
    int e = (int)(tid >> 5);
    int c = (int)(tid & 31) << 3;
    uint4 v = *(const uint4*)(ed + (size_t)e * 256 + c);
    const u16* u = (const u16*)&v;
    int s = snd[e], r = rcv[e];
    float* ps = sent + (size_t)s * 256 + c;
    float* pr = recv + (size_t)r * 256 + c;
#pragma unroll
    for (int i = 0; i < 8; ++i) {
        float f = bf2f(u[i]);
        atomicAdd(ps + i, f);
        atomicAdd(pr + i, f);
    }
}

__global__ void segsum_kernel(const u16* __restrict__ x, float* __restrict__ out, int per_g)
{
    int g = blockIdx.x, c = threadIdx.x, part = blockIdx.y, nparts = gridDim.y;
    int chunk = per_g / nparts;
    const u16* p = x + ((size_t)g * per_g + (size_t)part * chunk) * 256 + c;
    float s = 0.f;
    for (int i = 0; i < chunk; ++i) s += bf2f(p[(size_t)i * 256]);
    atomicAdd(out + (size_t)g * 256 + c, s);
}

// ---------------------------------------------------------------------------
extern "C" void kernel_launch(void* const* d_in, const int* in_sizes, int n_in,
                              void* d_out, int out_size, void* d_ws, size_t ws_size,
                              hipStream_t stream)
{
    (void)n_in; (void)out_size; (void)ws_size;
    const int G = 128, NPG = 512, EPG = 2048;
    const int N = G * NPG, E = G * EPG;

    // ---- resolve input layout (insertion order vs pytree-sorted) ----
    const float *in_nodes, *in_edges, *n_emb_w, *n_emb_b, *e_emb_w, *e_emb_b, *dw, *db;
    const int *snd, *rcv;
    const float *W[5][3], *Bv[5][3], *Gm[5][3], *Bt[5][3];  // mlp 0=edge,1=node,2=glob

    auto Fp = [&](int i) { return (const float*)d_in[i]; };
    auto Ip = [&](int i) { return (const int*)d_in[i]; };

    if (in_sizes[0] == N * 9) {   // top-level dict in insertion order
        in_nodes = Fp(0); in_edges = Fp(1); snd = Ip(2); rcv = Ip(3);
        n_emb_w = Fp(4); n_emb_b = Fp(5); e_emb_w = Fp(6); e_emb_b = Fp(7);
        if (in_sizes[8] >= 160000) {
            // gn_params in insertion order: per step edge(w,b,g,beta), node, glob
            for (int st = 0; st < 5; ++st)
                for (int m = 0; m < 3; ++m) {
                    int base = 8 + st * 12 + m * 4;
                    W[st][m] = Fp(base);     Bv[st][m] = Fp(base + 1);
                    Gm[st][m] = Fp(base + 2); Bt[st][m] = Fp(base + 3);
                }
        } else {
            // sorted keys within: per step edge(b,beta,g,w), glob, node
            const int mm[3] = {0, 2, 1};
            for (int st = 0; st < 5; ++st)
                for (int j = 0; j < 3; ++j) {
                    int m = mm[j], base = 8 + st * 12 + j * 4;
                    Bv[st][m] = Fp(base);     Bt[st][m] = Fp(base + 1);
                    Gm[st][m] = Fp(base + 2); W[st][m]  = Fp(base + 3);
                }
        }
        dw = Fp(68); db = Fp(69);
    } else {
        // full jax tree_flatten (alphabetical top-level)
        db = Fp(0); dw = Fp(1); e_emb_b = Fp(2); e_emb_w = Fp(3); in_edges = Fp(4);
        const int mm[3] = {0, 2, 1};
        for (int st = 0; st < 5; ++st)
            for (int j = 0; j < 3; ++j) {
                int m = mm[j], base = 5 + st * 12 + j * 4;
                Bv[st][m] = Fp(base);     Bt[st][m] = Fp(base + 1);
                Gm[st][m] = Fp(base + 2); W[st][m]  = Fp(base + 3);
            }
        n_emb_b = Fp(65); n_emb_w = Fp(66); in_nodes = Fp(67);
        rcv = Ip(68); snd = Ip(69);
    }

    // ---- carve workspace ----
    char* wsb = (char*)d_ws;
    size_t off = 0;
    auto carve = [&](size_t bytes) -> char* {
        char* p = wsb + off;
        off += (bytes + 255) & ~(size_t)255;
        return p;
    };
    u16* nodesA = (u16*)carve((size_t)N * 256 * 2);
    u16* nodesB = (u16*)carve((size_t)N * 256 * 2);
    u16* edgesA = (u16*)carve((size_t)E * 256 * 2);
    u16* edgesB = (u16*)carve((size_t)E * 256 * 2);
    float* sent = (float*)carve((size_t)N * 256 * 4);
    float* recv = (float*)carve((size_t)N * 256 * 4);
    float* globA = (float*)carve((size_t)G * 256 * 4);
    float* globB = (float*)carve((size_t)G * 256 * 4);
    float* n_attr = (float*)carve((size_t)G * 256 * 4);
    float* e_attr = (float*)carve((size_t)G * 256 * 4);
    u16 *wtE[5], *wtN[5], *wtG[5];
    for (int st = 0; st < 5; ++st) {
        wtE[st] = (u16*)carve((size_t)256 * 1024 * 2);
        wtN[st] = (u16*)carve((size_t)256 * 1024 * 2);
        wtG[st] = (u16*)carve((size_t)256 * 768 * 2);
    }
    u16* wtD = (u16*)carve((size_t)128 * 256 * 2);

    // ---- weight prep (transpose to N-major bf16) ----
    for (int st = 0; st < 5; ++st) {
        int ind = (st == 0) ? 896 : 1024;
        int gin = (st == 0) ? 640 : 768;
        prep_wt_kernel<<<(256 * ind + 255) / 256, 256, 0, stream>>>(W[st][0], wtE[st], ind, 256);
        prep_wt_kernel<<<(256 * ind + 255) / 256, 256, 0, stream>>>(W[st][1], wtN[st], ind, 256);
        prep_wt_kernel<<<(256 * gin + 255) / 256, 256, 0, stream>>>(W[st][2], wtG[st], gin, 256);
    }
    prep_wt_kernel<<<(128 * 256 + 255) / 256, 256, 0, stream>>>(dw, wtD, 256, 128);

    // ---- embeddings ----
    embed_kernel<<<(unsigned)((size_t)N * 256 / 256), 256, 0, stream>>>(in_nodes, n_emb_w, n_emb_b, nodesA, 9);
    embed_kernel<<<(unsigned)((size_t)E * 256 / 256), 256, 0, stream>>>(in_edges, e_emb_w, e_emb_b, edgesA, 3);
    zero_f32<<<(G * 256 + 255) / 256, 256, 0, stream>>>(globA, (size_t)G * 256);

    // ---- 5 GN steps ----
    u16 *nd = nodesA, *ndN = nodesB, *eg = edgesA, *egN = edgesB;
    float *gb = globA, *gbN = globB;
    for (int st = 0; st < 5; ++st) {
        int ind  = (st == 0) ? 896 : 1024;
        int gdim = (st == 0) ? 128 : 256;
        int gin  = (st == 0) ? 640 : 768;

        // edge update (fused gather GEMM)
        gn_mlp_wmma<0><<<E / 64, 256, 0, stream>>>(ind, gdim, 11,
            eg, nd, snd, rcv, nullptr, nullptr, gb,
            wtE[st], Bv[st][0], Gm[st][0], Bt[st][0], egN, nullptr);

        // scatter-add new edges into sent/recv (fp32 atomics)
        zero_f32<<<(unsigned)(((size_t)N * 256 + 255) / 256), 256, 0, stream>>>(sent, (size_t)N * 256);
        zero_f32<<<(unsigned)(((size_t)N * 256 + 255) / 256), 256, 0, stream>>>(recv, (size_t)N * 256);
        scatter_kernel<<<(unsigned)((size_t)E * 32 / 256), 256, 0, stream>>>(egN, snd, rcv, sent, recv);

        // node update
        gn_mlp_wmma<1><<<N / 64, 256, 0, stream>>>(ind, gdim, 9,
            nd, nullptr, nullptr, nullptr, sent, recv, gb,
            wtN[st], Bv[st][1], Gm[st][1], Bt[st][1], ndN, nullptr);

        // per-graph sums + global update
        zero_f32<<<(G * 256 + 255) / 256, 256, 0, stream>>>(n_attr, (size_t)G * 256);
        zero_f32<<<(G * 256 + 255) / 256, 256, 0, stream>>>(e_attr, (size_t)G * 256);
        segsum_kernel<<<dim3(G, 8),  256, 0, stream>>>(ndN, n_attr, NPG);
        segsum_kernel<<<dim3(G, 16), 256, 0, stream>>>(egN, e_attr, EPG);
        gn_mlp_wmma<2><<<G / 64, 256, 0, stream>>>(gin, gdim, 0,
            nullptr, nullptr, nullptr, nullptr, n_attr, e_attr, gb,
            wtG[st], Bv[st][2], Gm[st][2], Bt[st][2], nullptr, gbN);

        u16* t1 = nd; nd = ndN; ndN = t1;
        u16* t2 = eg; eg = egN; egN = t2;
        float* t3 = gb; gb = gbN; gbN = t3;
    }

    // ---- decoder ----
    dec_wmma<<<G / 16, 32, 0, stream>>>(gb, wtD, db, (float*)d_out);
}
